// Head_27324581937162
// MI455X (gfx1250) — matrix operand order repaired and checked by reading, not compiled, not run on hardware
//
#include <hip/hip_runtime.h>
#include <math.h>

// Single-head causal attention for MI455X (gfx1250), bf16 WMMA path.
// d_ws layout: q | k (bf16 [B*T,H]) | v transposed (bf16 [B][H][T]) | Wt (bf16 [3][H][C])
//   3 * 4 MB + 288 KB ~= 12.3 MB scratch.

typedef __attribute__((ext_vector_type(16))) __bf16 v16bf;
typedef __attribute__((ext_vector_type(8)))  __bf16 bf16x8;
typedef __attribute__((ext_vector_type(8)))  float  v8f;

#define B_  16
#define T_  2048
#define C_  768
#define H_  64
#define SCALE 0.03608439182435161f   // 1/sqrt(768) (reference scales by C**-0.5)

static __device__ __forceinline__ float half16_max(float v) {
#pragma unroll
    for (int m = 1; m <= 8; m <<= 1) v = fmaxf(v, __shfl_xor(v, m, 32));
    return v;
}
static __device__ __forceinline__ float half16_sum(float v) {
#pragma unroll
    for (int m = 1; m <= 8; m <<= 1) v += __shfl_xor(v, m, 32);
    return v;
}
// Load a 16-element bf16 fragment from two contiguous 8-element (b128) chunks.
static __device__ __forceinline__ void load16(v16bf& f, const __bf16* p0, const __bf16* p1) {
    const bf16x8 a = *reinterpret_cast<const bf16x8*>(p0);
    const bf16x8 b = *reinterpret_cast<const bf16x8*>(p1);
#pragma unroll
    for (int i = 0; i < 8; ++i) { f[i] = a[i]; f[i + 8] = b[i]; }
}

// ---------------------------------------------------------------------------
// Kernel 0: transpose+convert Wq|Wk|Wv f32 [C,H] -> Wt bf16 [3][H][C].
// ---------------------------------------------------------------------------
__global__ __launch_bounds__(256) void wt_convert_kernel(
    const float* __restrict__ Wq, const float* __restrict__ Wk,
    const float* __restrict__ Wv, __bf16* __restrict__ wt)
{
    const int idx = blockIdx.x * 256 + threadIdx.x;     // [3*H*C)
    if (idx >= 3 * H_ * C_) return;
    const int w = idx / (H_ * C_);
    const int r = idx - w * (H_ * C_);
    const int h = r / C_;
    const int c = r - h * C_;
    const float* W = (w == 0) ? Wq : (w == 1) ? Wk : Wv;
    wt[idx] = (__bf16)W[(size_t)c * H_ + h];
}

// ---------------------------------------------------------------------------
// Kernel 1: fused QKV projection.  [B*T, C] @ [C, 64]x3, f32 -> bf16 WMMA.
// Block = 256 threads = 8 waves; each wave owns a 16-row strip and all 12
// column tiles. Per K-step, ALL 12 B fragments are preloaded; a
// sched_barrier pins the load burst ahead of the WMMA drain so each WMMA
// waits on a descending loadcnt instead of 0.
// ---------------------------------------------------------------------------
__global__ __launch_bounds__(256) void qkv_proj_kernel(
    const float* __restrict__ x, const __bf16* __restrict__ wt,
    __bf16* __restrict__ qb, __bf16* __restrict__ kb, __bf16* __restrict__ vt)
{
    const int lane = threadIdx.x & 31;
    const int wave = threadIdx.x >> 5;          // 0..7
    const int hi   = lane >> 4;
    const int ln   = lane & 15;
    const int rowBase = blockIdx.x * 128 + wave * 16;   // 16-row strip

    v8f acc[12];
#pragma unroll
    for (int i = 0; i < 12; ++i)
#pragma unroll
        for (int j = 0; j < 8; ++j) acc[i][j] = 0.0f;

    const float* xrow0 = x + (size_t)(rowBase + ln) * C_;

    for (int kk = 0; kk < C_; kk += 32) {
        // A fragment: rows rowBase+ln (M = lane%16), two contiguous K groups.
        v16bf a;
        {
            const float* xr = xrow0 + kk;
            const float4 g0a = *reinterpret_cast<const float4*>(xr + 8 * hi);
            const float4 g0b = *reinterpret_cast<const float4*>(xr + 8 * hi + 4);
            const float4 g1a = *reinterpret_cast<const float4*>(xr + 16 + 8 * hi);
            const float4 g1b = *reinterpret_cast<const float4*>(xr + 16 + 8 * hi + 4);
            a[0]=(__bf16)g0a.x; a[1]=(__bf16)g0a.y; a[2]=(__bf16)g0a.z; a[3]=(__bf16)g0a.w;
            a[4]=(__bf16)g0b.x; a[5]=(__bf16)g0b.y; a[6]=(__bf16)g0b.z; a[7]=(__bf16)g0b.w;
            a[8]=(__bf16)g1a.x; a[9]=(__bf16)g1a.y; a[10]=(__bf16)g1a.z; a[11]=(__bf16)g1a.w;
            a[12]=(__bf16)g1b.x;a[13]=(__bf16)g1b.y;a[14]=(__bf16)g1b.z; a[15]=(__bf16)g1b.w;
        }

        // Preload ALL 12 B fragments (24 x global_load_b128 burst).
        v16bf bfr[12];
#pragma unroll
        for (int ct = 0; ct < 12; ++ct) {
            const int w  = ct >> 2;
            const int n0 = (ct & 3) * 16 + ln;           // N = lane%16
            const __bf16* p = wt + ((size_t)(w * H_ + n0) * C_) + kk + 8 * hi;
            load16(bfr[ct], p, p + 16);
        }
        // Pin the burst ahead of the drain: nothing crosses this point.
        __builtin_amdgcn_sched_barrier(0);

        // Drain with 12 back-to-back WMMAs; waits count down across the burst.
#pragma unroll
        for (int ct = 0; ct < 12; ++ct)
            acc[ct] = __builtin_amdgcn_wmma_f32_16x16x32_bf16(
                false, a, false, bfr[ct], (short)0, acc[ct], false, false);
    }

    // C/D layout: row M = r + 8*hi, col N = lane%16
#pragma unroll
    for (int ct = 0; ct < 8; ++ct) {                    // q and k: row-major [B*T,H]
        __bf16* dst = (ct < 4) ? qb : kb;
        const int n = (ct & 3) * 16 + ln;
#pragma unroll
        for (int r = 0; r < 8; ++r) {
            const int row = rowBase + r + 8 * hi;
            dst[(size_t)row * H_ + n] = (__bf16)acc[ct][r];
        }
    }
#pragma unroll
    for (int ct = 8; ct < 12; ++ct) {                   // v: transposed [B][H][T]
        const int n = (ct & 3) * 16 + ln;
#pragma unroll
        for (int r = 0; r < 8; ++r) {
            const int row = rowBase + r + 8 * hi;
            const int b   = row >> 11;                   // /T_
            const int t   = row & (T_ - 1);
            vt[((size_t)b * H_ + n) * T_ + t] = (__bf16)acc[ct][r];
        }
    }
}

// ---------------------------------------------------------------------------
// Kernel 2: flash attention with online softmax.
// Grid = (T/64, B); block = 128 threads = 4 waves. Wave w owns query rows
// [t0+16w, t0+16w+16) x all 64 head dims (4 f32 accumulator tiles).
// sched_barriers keep the K/V load bursts ahead of dependent compute.
// ---------------------------------------------------------------------------
__global__ __launch_bounds__(128) void flash_attn_kernel(
    const __bf16* __restrict__ qb, const __bf16* __restrict__ kb,
    const __bf16* __restrict__ vt, float* __restrict__ out)
{
    __shared__ __bf16 pbuf[4][16][32];   // per-wave 16x32 P tile (4 KB)

    const int lane = threadIdx.x & 31;
    const int wave = threadIdx.x >> 5;   // 0..3
    const int hi   = lane >> 4;
    const int ln   = lane & 15;
    const int b    = blockIdx.y;
    const int t0   = blockIdx.x * 64;
    const int tq0  = t0 + wave * 16;
    const float NEG_INF = -__builtin_inff();

    // Q fragments held in registers for the whole key loop (2 K-chunks of 32).
    v16bf qf[2];
    {
        const __bf16* qrow = qb + ((size_t)b * T_ + tq0 + ln) * H_;
#pragma unroll
        for (int hk = 0; hk < 2; ++hk)
            load16(qf[hk], qrow + hk * 32 + 8 * hi, qrow + hk * 32 + 16 + 8 * hi);
    }

    float mstate[8], lstate[8];
    v8f acc[4];
#pragma unroll
    for (int r = 0; r < 8; ++r) { mstate[r] = NEG_INF; lstate[r] = 0.0f; }
#pragma unroll
    for (int ht = 0; ht < 4; ++ht)
#pragma unroll
        for (int r = 0; r < 8; ++r) acc[ht][r] = 0.0f;

    const __bf16* kbase = kb + (size_t)b * T_ * H_;
    const __bf16* vbase = vt + (size_t)b * H_ * T_;

    // causal: only key blocks with s0 <= t0+63 contribute (uniform per block)
    for (int s0 = 0; s0 < t0 + 64; s0 += 32) {
        // Prefetch next key block (global_prefetch_b8 hint).
        if (s0 + 32 < t0 + 64) {
            __builtin_prefetch(kbase + (size_t)(s0 + 32 + lane) * H_, 0, 1);
            __builtin_prefetch(vbase + (size_t)(lane * 2) * T_ + s0 + 32, 0, 1);
        }

        // ---- load all K fragments (8 x b128 burst), then 4 S WMMAs ----
        v16bf kf[2][2];                  // [nt][hk]
#pragma unroll
        for (int nt = 0; nt < 2; ++nt) {
            const __bf16* krow = kbase + (size_t)(s0 + nt * 16 + ln) * H_;
#pragma unroll
            for (int hk = 0; hk < 2; ++hk)
                load16(kf[nt][hk], krow + hk * 32 + 8 * hi,
                                   krow + hk * 32 + 16 + 8 * hi);
        }
        __builtin_amdgcn_sched_barrier(0);   // keep the K burst ahead of WMMAs

        v8f S[2];
#pragma unroll
        for (int nt = 0; nt < 2; ++nt) {
            v8f sacc;
#pragma unroll
            for (int j = 0; j < 8; ++j) sacc[j] = 0.0f;
#pragma unroll
            for (int hk = 0; hk < 2; ++hk)
                sacc = __builtin_amdgcn_wmma_f32_16x16x32_bf16(
                    false, qf[hk], false, kf[nt][hk], (short)0, sacc, false, false);
            S[nt] = sacc;
        }

        // ---- issue V-fragment loads now; softmax VALU hides their latency ----
        v16bf vf[4];
#pragma unroll
        for (int ht = 0; ht < 4; ++ht) {
            const __bf16* vrow = vbase + (size_t)(ht * 16 + ln) * T_ + s0 + 8 * hi;
            load16(vf[ht], vrow, vrow + 16);
        }
        __builtin_amdgcn_sched_barrier(0);   // don't sink V loads past softmax

        // ---- scale + causal mask ----
#pragma unroll
        for (int nt = 0; nt < 2; ++nt)
#pragma unroll
            for (int r = 0; r < 8; ++r) {
                const int t = tq0 + r + 8 * hi;
                const int s = s0 + nt * 16 + ln;
                const float v = S[nt][r] * SCALE;
                S[nt][r] = (s <= t) ? v : NEG_INF;
            }

        // ---- online softmax (row r+8*hi lives across 16 lanes) ----
#pragma unroll
        for (int r = 0; r < 8; ++r) {
            const float rowmax = half16_max(fmaxf(S[0][r], S[1][r]));
            const float mnew   = fmaxf(mstate[r], rowmax);
            const float alpha  = __expf(mstate[r] - mnew);
            const float p0 = __expf(S[0][r] - mnew);
            const float p1 = __expf(S[1][r] - mnew);
            S[0][r] = p0; S[1][r] = p1;
            const float rsum = half16_sum(p0 + p1);
            lstate[r] = lstate[r] * alpha + rsum;
            mstate[r] = mnew;
#pragma unroll
            for (int ht = 0; ht < 4; ++ht) acc[ht][r] *= alpha;
        }

        // ---- stage P (C/D layout) to LDS, reload as A fragment ----
        __bf16 (*pl)[32] = pbuf[wave];
#pragma unroll
        for (int nt = 0; nt < 2; ++nt)
#pragma unroll
            for (int r = 0; r < 8; ++r)
                pl[r + 8 * hi][nt * 16 + ln] = (__bf16)S[nt][r];
        __syncthreads();

        v16bf pf;                         // A frag: two contiguous K groups
        load16(pf, &pl[ln][8 * hi], &pl[ln][16 + 8 * hi]);

        // ---- O += P V  (K = 32 keys, 4 N tiles of 16 head dims) ----
#pragma unroll
        for (int ht = 0; ht < 4; ++ht)
            acc[ht] = __builtin_amdgcn_wmma_f32_16x16x32_bf16(
                false, pf, false, vf[ht], (short)0, acc[ht], false, false);
        __syncthreads();
    }

    // ---- epilogue: out = acc / l ----
#pragma unroll
    for (int ht = 0; ht < 4; ++ht)
#pragma unroll
        for (int r = 0; r < 8; ++r) {
            const int t = tq0 + r + 8 * hi;
            out[((size_t)b * T_ + t) * H_ + ht * 16 + ln] = acc[ht][r] / lstate[r];
        }
}

// ---------------------------------------------------------------------------
extern "C" void kernel_launch(void* const* d_in, const int* in_sizes, int n_in,
                              void* d_out, int out_size, void* d_ws, size_t ws_size,
                              hipStream_t stream) {
    (void)in_sizes; (void)n_in; (void)out_size; (void)ws_size;
    const float* x  = (const float*)d_in[0];
    const float* Wq = (const float*)d_in[1];
    const float* Wk = (const float*)d_in[2];
    const float* Wv = (const float*)d_in[3];
    float* out = (float*)d_out;

    const size_t qkvElems = (size_t)B_ * T_ * H_;       // 2,097,152
    __bf16* qb = (__bf16*)d_ws;                         // 4 MB
    __bf16* kb = qb + qkvElems;                         // 4 MB
    __bf16* vt = kb + qkvElems;                         // 4 MB (transposed [B][H][T])
    __bf16* wt = vt + qkvElems;                         // 288 KB ([3][H][C])

    // 0) W transpose+convert to bf16
    wt_convert_kernel<<<dim3((3 * H_ * C_ + 255) / 256), 256, 0, stream>>>(
        Wq, Wk, Wv, wt);

    // 1) QKV projection: 32768 rows / 128 rows per block
    qkv_proj_kernel<<<dim3((B_ * T_) / 128), 256, 0, stream>>>(
        x, wt, qb, kb, vt);

    // 2) Flash attention: (T/64) row-blocks x B batches, 4 waves each
    flash_attn_kernel<<<dim3(T_ / 64, B_), 128, 0, stream>>>(qb, kb, vt, out);
}
